// VQ_43130061586925
// MI455X (gfx1250) — matrix operand order repaired and checked by reading, not compiled, not run on hardware
//
#include <hip/hip_runtime.h>
#include <hip/hip_bf16.h>

typedef __attribute__((ext_vector_type(16))) __bf16 v16bf;
typedef __attribute__((ext_vector_type(8)))  float  v8f;

#define NCODES 512
#define DIM    64
#define HW     4096      // 64*64
#define CHW    262144    // 64*4096
// ws layout (bytes):
//   [0, 65536)      : codebook bf16, B-matrix layout (32 tiles x 2 kchunks x 32 lanes x 16 elems)
//   [65536, 67584)  : cnorm[512] f32
//   [67584, 75776)  : partials[2048] f32

// --- repack codebook f32 -> bf16 in WMMA B-matrix (32x16, K x N) lane layout ---
// B k order used here (contiguous, same as A; not AMD's table order): lanes 0-15 hold K=0..15,
// lanes 16-31 hold K=16..31.  => element e of v16bf at lane l: k_local = 16*(l>>4) + e.
__global__ void vq_prep_b(const float* __restrict__ cb, __bf16* __restrict__ wsb) {
    int tid = blockIdx.x * 256 + threadIdx.x;   // 0..32767
    int e    = tid & 15;
    int lane = (tid >> 4) & 31;
    int kk   = (tid >> 9) & 1;                  // K chunk (0..1), 32 K each
    int ct   = tid >> 10;                       // code tile (0..31), 16 codes each
    int n     = lane & 15;
    int khalf = lane >> 4;
    int k     = kk * 32 + khalf * 16 + e;
    int code  = ct * 16 + n;
    wsb[tid] = (__bf16)cb[code * DIM + k];
}

__global__ void vq_prep_cnorm(const float* __restrict__ cb, float* __restrict__ cnorm) {
    int c = blockIdx.x * 256 + threadIdx.x;
    if (c < NCODES) {
        float s = 0.f;
        #pragma unroll
        for (int k = 0; k < DIM; ++k) { float v = cb[c * DIM + k]; s = fmaf(v, v, s); }
        cnorm[c] = s;
    }
}

// --- main: one workgroup per (b,h): 64 spatial positions x 512 codes ---
__launch_bounds__(256)
__global__ void vq_main(const float* __restrict__ x, const float* __restrict__ cb,
                        const __bf16* __restrict__ wsb, const float* __restrict__ cnorm,
                        float* __restrict__ outq, float* __restrict__ partials) {
    // A-matrix staging: 4 rowgroups x 2 kchunks x 32 lanes x 16 bf16 = 4096
    __shared__ __align__(32) __bf16 lds_a[4096];
    __shared__ float lds_xf[4096];      // f32 copy, [d*64 + w]
    __shared__ float red_val[2048];     // 8 waves x 8 vgprs x 32 lanes
    __shared__ int   red_idx[2048];
    __shared__ int   lds_idx[64];

    const int t = threadIdx.x;
    const int b = blockIdx.x >> 6;
    const int h = blockIdx.x & 63;
    const long base = (long)b * CHW + (long)h * 64;

    // Stage x tile: coalesced 64-float runs per channel d (NCHW: w contiguous).
    // Scatter into LDS in the A-matrix (16x32 bf16) layout:
    //   lane = 16*khalf + (w&15); element e: lanes 0-15 hold K 0-15,
    //   lanes 16-31 hold K 16-31.  khalf=kd>>4, e=kd&15.
    for (int i = 0; i < 16; ++i) {
        int id = i * 256 + t;
        int d = id >> 6, w = id & 63;
        float v = x[base + (long)d * HW + w];
        lds_xf[d * 64 + w] = v;
        int kk = d >> 5, kd = d & 31;
        int khalf = kd >> 4;
        int e = kd & 15;
        int r = w >> 4, m = w & 15;
        lds_a[((r * 2 + kk) * 32 + (khalf * 16 + m)) * 16 + e] = (__bf16)v;
    }
    __syncthreads();

    const int wv = t >> 5;          // wave id 0..7
    const int l  = t & 31;          // lane
    const int r  = wv >> 1;         // row group (16 rows)
    const int g  = wv & 1;          // code group (256 codes)

    v16bf a0 = *(const v16bf*)&lds_a[((r * 2 + 0) * 32 + l) * 16];
    v16bf a1 = *(const v16bf*)&lds_a[((r * 2 + 1) * 32 + l) * 16];

    float best[8];
    int   bidx[8];
    #pragma unroll
    for (int v = 0; v < 8; ++v) { best[v] = 3.4e38f; bidx[v] = 0; }

    for (int ct = 0; ct < 16; ++ct) {
        int ctg = g * 16 + ct;
        v16bf b0 = *(const v16bf*)(wsb + ((ctg * 2 + 0) * 32 + l) * 16);
        v16bf b1 = *(const v16bf*)(wsb + ((ctg * 2 + 1) * 32 + l) * 16);
        v8f acc = {0.f, 0.f, 0.f, 0.f, 0.f, 0.f, 0.f, 0.f};
        acc = __builtin_amdgcn_wmma_f32_16x16x32_bf16(false, a0, false, b0,
                                                      (short)0, acc, false, false);
        acc = __builtin_amdgcn_wmma_f32_16x16x32_bf16(false, a1, false, b1,
                                                      (short)0, acc, false, false);
        int code = g * 256 + ct * 16 + (l & 15);
        float cn = cnorm[code];
        #pragma unroll
        for (int v = 0; v < 8; ++v) {
            // relative distance: ||c||^2 - 2 x.c   (||x||^2 constant per row)
            float d2 = fmaf(-2.0f, acc[v], cn);
            if (d2 < best[v]) { best[v] = d2; bidx[v] = code; }   // ct ascending => first-min
        }
    }
    #pragma unroll
    for (int v = 0; v < 8; ++v) {
        red_val[wv * 256 + v * 32 + l] = best[v];
        red_idx[wv * 256 + v * 32 + l] = bidx[v];
    }
    __syncthreads();

    // per-row argmin across 16 lanes (N dim) and 2 code groups
    if (t < 64) {
        int row = t;
        int rr = row >> 4, local = row & 15;
        int khalf = local >> 3, v = local & 7;   // C/D layout: M = v + 8*khalf
        float bv = 3.4e38f; int bi = 0x7fffffff;
        for (int gg = 0; gg < 2; ++gg) {
            int bo = (rr * 2 + gg) * 256 + v * 32 + khalf * 16;
            for (int j = 0; j < 16; ++j) {
                float cv = red_val[bo + j];
                int   ci = red_idx[bo + j];
                if (cv < bv || (cv == bv && ci < bi)) { bv = cv; bi = ci; }
            }
        }
        lds_idx[row] = bi;
    }
    __syncthreads();

    // output (NCHW coalesced) + loss partial, all in f32
    float lsum = 0.f;
    for (int i = 0; i < 16; ++i) {
        int id = i * 256 + t;
        int d = id >> 6, w = id & 63;
        float cv = cb[lds_idx[w] * DIM + d];
        float xv = lds_xf[d * 64 + w];
        float diff = cv - xv;
        lsum = fmaf(diff, diff, lsum);
        outq[base + (long)d * HW + w] = xv + (cv - xv);   // STE value
    }
    __syncthreads();
    red_val[t] = lsum;
    __syncthreads();
    for (int s = 128; s > 0; s >>= 1) {
        if (t < s) red_val[t] += red_val[t + s];
        __syncthreads();
    }
    if (t == 0) partials[blockIdx.x] = red_val[0];
}

__global__ void vq_finish(const float* __restrict__ partials, float* __restrict__ loss) {
    __shared__ float s[256];
    int t = threadIdx.x;
    float sum = 0.f;
    for (int i = t; i < 2048; i += 256) sum += partials[i];   // fixed order: deterministic
    s[t] = sum;
    __syncthreads();
    for (int k = 128; k > 0; k >>= 1) { if (t < k) s[t] += s[t + k]; __syncthreads(); }
    if (t == 0) loss[0] = s[0] * (1.25f / 8388608.0f);  // (1 + 0.25) * mean
}

extern "C" void kernel_launch(void* const* d_in, const int* in_sizes, int n_in,
                              void* d_out, int out_size, void* d_ws, size_t ws_size,
                              hipStream_t stream) {
    (void)in_sizes; (void)n_in; (void)out_size; (void)ws_size;
    const float* x  = (const float*)d_in[0];   // [32,64,64,64] NCHW f32
    const float* cb = (const float*)d_in[1];   // [512,64] f32
    float* out = (float*)d_out;                // [0] loss, [1..] quantized NCHW
    char* ws = (char*)d_ws;
    __bf16* wsb     = (__bf16*)ws;
    float* cnorm    = (float*)(ws + 65536);
    float* partials = (float*)(ws + 67584);

    vq_prep_b<<<128, 256, 0, stream>>>(cb, wsb);
    vq_prep_cnorm<<<2, 256, 0, stream>>>(cb, cnorm);
    vq_main<<<2048, 256, 0, stream>>>(x, cb, wsb, cnorm, out + 1, partials);
    vq_finish<<<1, 256, 0, stream>>>(partials, out);
}
